// DECLayer_7756710936991
// MI455X (gfx1250) — matrix operand order repaired and checked by reading, not compiled, not run on hardware
//
#include <hip/hip_runtime.h>

// DEC soft-assignment layer for MI455X (gfx1250, wave32).
// q[n,k] = (1/(1+d2)) / rowsum,  d2 = ||x||^2 + ||c||^2 - 2 x.c   (alpha==1 -> exponent 1)
// Strategy: exact f32 norms + f16 WMMA (f32 accumulate) for the cross term.
// HBM traffic ~537MB -> ~23us floor; f16 WMMA keeps compute under that floor.
// Staging uses CDNA5 async global->LDS copies (ASYNCcnt) to avoid the
// load->waitcnt->ds_store serialization the generic path produced.

typedef __attribute__((ext_vector_type(16))) _Float16 v16h;
typedef __attribute__((ext_vector_type(8)))  float    v8f;
typedef __attribute__((ext_vector_type(4)))  int      v4i;   // b128 packet type for async copies

#define DDIM    512              // feature dim
#define KCL     512              // number of clusters (output columns)
#define TM      64               // rows per block
#define DK      32               // K-depth per chunk (one WMMA K)
#define NCHUNK  (DDIM / DK)      // 16
#define NTILE   (KCL / 16)       // 32 column tiles
#define XSTRIDE 36               // padded f32 row stride in LDS (bank-conflict avoidance)
#define THREADS 512              // 16 wave32s

// Generic-pointer cast for the gfx1250 async-copy builtins (they take int4* params).
#define B128P(p) ((v4i*)(void*)(p))

// Workspace layout (d_ws):
//   [0,    2KB)        : cnorm[512]  (f32  ||c_k||^2)
//   [4KB,  4KB+512KB)  : Bp — clusters as f16, pre-packed in WMMA B-fragment order:
//                        Bp[kc][nt][lane][h] = (f16) C[nt*16 + (lane&15)][kc*32 + (lane>>4)*16 + h]

__global__ void dec_pack_clusters(const float* __restrict__ C, _Float16* __restrict__ Bp) {
    int idx = blockIdx.x * blockDim.x + threadIdx.x;   // 16*32*32 = 16384 threads
    int L   = idx & 31;
    int nt  = (idx >> 5) & 31;
    int kc  = idx >> 10;
    int n   = nt * 16 + (L & 15);
    int kb  = kc * 32 + ((L >> 4) << 4);
    const float* src = C + (size_t)n * DDIM + kb;
    _Float16* dst = Bp + (size_t)idx * 16;
#pragma unroll
    for (int h = 0; h < 16; ++h) dst[h] = (_Float16)src[h];
}

__global__ void dec_cluster_norms(const float* __restrict__ C, float* __restrict__ cn) {
    int k = threadIdx.x;                               // 512 threads, one per cluster
    const float4* row = (const float4*)(C + (size_t)k * DDIM);
    float s = 0.f;
#pragma unroll 4
    for (int j = 0; j < DDIM / 4; ++j) {
        float4 v = row[j];
        s = fmaf(v.x, v.x, s); s = fmaf(v.y, v.y, s);
        s = fmaf(v.z, v.z, s); s = fmaf(v.w, v.w, s);
    }
    cn[k] = s;
}

__launch_bounds__(THREADS)
__global__ void dec_main(const float* __restrict__ X,
                         const _Float16* __restrict__ Bp,
                         const float* __restrict__ cn,
                         float* __restrict__ Out) {
    __shared__ __align__(16) float    xs[TM * XSTRIDE];        // 9216 B
    __shared__ __align__(16) _Float16 bs[NTILE * 32 * 16];     // 32768 B (one B chunk)
    __shared__ float rowsum[TM];
    __shared__ float x2s[TM];

    const int tid  = threadIdx.x;
    const int lane = tid & 31;         // wave32 on gfx1250
    const int wave = tid >> 5;         // 0..15
    const int mt   = wave & 3;         // M-tile (16 rows each)
    const int nt0  = (wave >> 2) * 8;  // first of 8 N-tiles for this wave
    const long rowBase = (long)blockIdx.x * TM;

    if (tid < TM) rowsum[tid] = 0.f;

    v8f acc[8];
#pragma unroll
    for (int j = 0; j < 8; ++j)
#pragma unroll
        for (int i = 0; i < 8; ++i) acc[j][i] = 0.f;

    float x2part = 0.f;
    const int rS = tid >> 3;           // staging row  (0..63)
    const int cS = (tid & 7) * 4;      // staging col  (float4 granularity)

    // Per-thread staging pointers (global src / LDS dst share identical layouts,
    // so the async-copy's shared INST_OFFSET serves both sides).
    const float*  xg  = X + rS * (long)DDIM + cS;            // + rowBase*DDIM added below
    float*        xl  = xs + rS * XSTRIDE + cS;
    const float4* bg0 = (const float4*)Bp + tid;             // + chunk offset below
    float4*       bl0 = (float4*)bs + tid;

    for (int kc = 0; kc < NCHUNK; ++kc) {
        __syncthreads();  // previous chunk's LDS consumers done (also covers rowsum init)

        // stage 64x32 f32 X tile: one async 16B packet per thread
        __builtin_amdgcn_global_load_async_to_lds_b128(
            B128P(xg + rowBase * DDIM + kc * DK), B128P(xl), 0, 0);

        // stage 32KB pre-packed B chunk: four async 16B packets per thread,
        // INST_OFFSET advances global and LDS addresses together.
        {
            const float4* bg = bg0 + (size_t)kc * (NTILE * 32 * 2);  // 2048 float4 per chunk
            __builtin_amdgcn_global_load_async_to_lds_b128(B128P(bg), B128P(bl0), 0,     0);
            __builtin_amdgcn_global_load_async_to_lds_b128(B128P(bg), B128P(bl0), 8192,  0);
            __builtin_amdgcn_global_load_async_to_lds_b128(B128P(bg), B128P(bl0), 16384, 0);
            __builtin_amdgcn_global_load_async_to_lds_b128(B128P(bg), B128P(bl0), 24576, 0);
        }

        __builtin_amdgcn_s_wait_asynccnt(0);
        __syncthreads();

        // ||x||^2 partials (threads 0..63, one row each) from staged tile
        if (tid < TM) {
            const float* xr = xs + tid * XSTRIDE;
#pragma unroll
            for (int j = 0; j < DK; ++j) x2part = fmaf(xr[j], xr[j], x2part);
        }

        // Build A fragment in contiguous K order (matches packed B):
        // lanes 0-15 (g=0): halves 0-15 -> K0..15
        // lanes 16-31(g=1): halves 0-15 -> K16..31
        const int ar = mt * 16 + (lane & 15);
        const int g8 = (lane >> 4) * 16;
        const float* p0 = xs + ar * XSTRIDE + g8;
        v16h a;
#pragma unroll
        for (int h = 0; h < 8; ++h) {
            a[h]     = (_Float16)p0[h];
            a[h + 8] = (_Float16)p0[h + 8];
        }

        // 8 WMMAs with a 2-deep B-fragment pipeline: the ds_load for tile j+1
        // is issued before the WMMA for tile j so DS latency overlaps the XDL pipe.
        v16h bcur = *(const v16h*)(bs + ((nt0 << 5) + lane) * 16);
#pragma unroll
        for (int j = 0; j < 8; ++j) {
            v16h bnext;
            if (j < 7)
                bnext = *(const v16h*)(bs + (((nt0 + j + 1) << 5) + lane) * 16);
            acc[j] = __builtin_amdgcn_wmma_f32_16x16x32_f16(
                false, a, false, bcur, (short)0, acc[j], false, false);
            bcur = bnext;
        }
    }

    if (tid < TM) x2s[tid] = x2part;
    __syncthreads();

    // Epilogue. C/D layout: lane L, element i -> row = i + 8*(L>>4), col = L&15.
    const int colL = lane & 15;
    const int rrow = mt * 16 + ((lane >> 4) << 3);

    float c2v[8];
#pragma unroll
    for (int j = 0; j < 8; ++j) c2v[j] = cn[(nt0 + j) * 16 + colL];

#pragma unroll
    for (int i = 0; i < 8; ++i) {
        float x2 = x2s[rrow + i];
        float s  = 0.f;
#pragma unroll
        for (int j = 0; j < 8; ++j) {
            float d2 = fmaxf(x2 + c2v[j] - 2.0f * acc[j][i], 0.0f);
            float q  = 1.0f / (1.0f + d2);
            acc[j][i] = q;
            s += q;
        }
        atomicAdd(&rowsum[rrow + i], s);   // ds_add_f32, block-local
    }
    __syncthreads();

#pragma unroll
    for (int i = 0; i < 8; ++i) {
        float inv = 1.0f / rowsum[rrow + i];
        long  orow = (rowBase + rrow + i) * (long)KCL;
#pragma unroll
        for (int j = 0; j < 8; ++j)
            Out[orow + (nt0 + j) * 16 + colL] = acc[j][i] * inv;
    }
}

extern "C" void kernel_launch(void* const* d_in, const int* in_sizes, int n_in,
                              void* d_out, int out_size, void* d_ws, size_t ws_size,
                              hipStream_t stream) {
    const float* X = (const float*)d_in[0];   // [N, 512] f32
    const float* C = (const float*)d_in[1];   // [512, 512] f32
    float* Out = (float*)d_out;               // [N, 512] f32
    (void)n_in; (void)out_size; (void)ws_size;

    const int N = in_sizes[0] / DDIM;         // 131072

    float*    cnorm = (float*)d_ws;
    _Float16* Bp    = (_Float16*)((char*)d_ws + 4096);

    dec_pack_clusters<<<32, 512, 0, stream>>>(C, Bp);          // 16384 threads
    dec_cluster_norms<<<1, 512, 0, stream>>>(C, cnorm);
    dec_main<<<N / TM, THREADS, 0, stream>>>(X, Bp, cnorm, Out);
}
